// LSTM_30176440221713
// MI455X (gfx1250) — matrix operand order repaired and checked by reading, not compiled, not run on hardware
//
#include <hip/hip_runtime.h>
#include <math.h>

// ---------------------------------------------------------------------------
// LSTM on MI455X (gfx1250): fused per-step GEMM (bf16 WMMA, f32 accum) + cell
//   gates_s = [x_s | h_{s-1}] @ [Wx | Wh]^T + b      (64 x 4096, K = 1536)
//   i,f,j,o -> c,h update fused in-register per wave.
// ---------------------------------------------------------------------------

#define SEQ   512
#define BATCH 64
#define DIN   512
#define HID   1024
#define G4    (4 * HID)

typedef __attribute__((ext_vector_type(16))) __bf16 v16bf;
typedef __attribute__((ext_vector_type(8)))  float  v8f;

union ABCast { uint4 q[2]; v16bf v; };

__device__ __forceinline__ unsigned short f2bf(float f) {
  // round-to-nearest-even f32 -> bf16
  unsigned u = __builtin_bit_cast(unsigned, f);
  u += 0x7FFFu + ((u >> 16) & 1u);
  return (unsigned short)(u >> 16);
}

__device__ __forceinline__ unsigned pack2(float a, float b) {
  return (unsigned)f2bf(a) | ((unsigned)f2bf(b) << 16);
}

__device__ __forceinline__ float sigm(float x) {
  return 1.0f / (1.0f + __expf(-x));
}

__device__ __forceinline__ v16bf load_b16(const unsigned short* __restrict__ p) {
  // 32 contiguous bytes of bf16 weights: column = lane&15, K-half = lane>>4
  ABCast u;
  u.q[0] = ((const uint4*)p)[0];
  u.q[1] = ((const uint4*)p)[1];
  return u.v;
}

// ---------------------------------------------------------------------------
// Prep: convert Wx/Wh to bf16 in workspace, seed h/c state buffers.
// ---------------------------------------------------------------------------
__global__ __launch_bounds__(256) void lstm_prep_kernel(
    const float* __restrict__ wx_f, const float* __restrict__ wh_f,
    const float* __restrict__ h0,   const float* __restrict__ c0,
    unsigned short* __restrict__ wx_b, unsigned short* __restrict__ wh_b,
    float* __restrict__ hbuf0, float* __restrict__ cbuf)
{
  const int i = blockIdx.x * 256 + threadIdx.x;
  if (i < G4 * DIN)  wx_b[i] = f2bf(wx_f[i]);
  if (i < G4 * HID)  wh_b[i] = f2bf(wh_f[i]);
  if (i < BATCH * HID) {
    hbuf0[i] = h0[i];
    cbuf[i]  = c0[i];
  }
}

// ---------------------------------------------------------------------------
// One LSTM step. Grid: (8, 4) blocks x 256 threads (8 waves).
//   blockIdx.y -> M-tile (16 batch rows), blockIdx.x*8 + wave -> h-tile (16 cols)
//   Each wave accumulates 4 gate tiles (i,f,j,o) over K=1536, then updates c,h.
// ---------------------------------------------------------------------------
#define LDS_STRIDE 1544   // 1536 bf16 + 8 pad -> bank spread for ds_load_b128

__global__ __launch_bounds__(256) void lstm_step_kernel(
    const float* __restrict__ xs,            // [64][512] step slice
    const unsigned short* __restrict__ wx,   // bf16 [4096][512]
    const unsigned short* __restrict__ wh,   // bf16 [4096][1024]
    const float* __restrict__ bias,          // [4096]
    const float* __restrict__ hprev,         // [64][1024]
    float* __restrict__ hnext,               // [64][1024]
    float* __restrict__ cst)                 // [64][1024] (in/out)
{
  __shared__ unsigned short Ap[16 * LDS_STRIDE];   // 48.25 KB: [x_s | h] bf16 panel

  const int tid    = threadIdx.x;
  const int m_base = blockIdx.y * 16;

  // ---- Stage A panel to LDS with f32->bf16 conversion ----
  // x part: 16 rows x 512 cols = 2048 float4
#pragma unroll
  for (int q = 0; q < 8; ++q) {
    const int e  = tid + 256 * q;
    const int r  = e >> 7;
    const int c4 = (e & 127) << 2;
    const float4 v = *(const float4*)(xs + (size_t)(m_base + r) * DIN + c4);
    *(uint2*)&Ap[r * LDS_STRIDE + c4] = make_uint2(pack2(v.x, v.y), pack2(v.z, v.w));
  }
  // h part: 16 rows x 1024 cols = 4096 float4
#pragma unroll
  for (int q = 0; q < 16; ++q) {
    const int e  = tid + 256 * q;
    const int r  = e >> 8;
    const int c4 = (e & 255) << 2;
    const float4 v = *(const float4*)(hprev + (size_t)(m_base + r) * HID + c4);
    *(uint2*)&Ap[r * LDS_STRIDE + DIN + c4] = make_uint2(pack2(v.x, v.y), pack2(v.z, v.w));
  }
  __syncthreads();

  const int w      = tid >> 5;
  const int lane   = tid & 31;
  const int h_base = (blockIdx.x * 8 + w) * 16;
  const int nl     = lane & 15;      // output column within tile / A row
  const int kh     = lane >> 4;      // K-half selector

  v8f acc0 = {}, acc1 = {}, acc2 = {}, acc3 = {};
  const unsigned short* Abase = &Ap[nl * LDS_STRIDE];

  // ---- K loop over input projection (Wx, K = 0..511) ----
  for (int kk = 0; kk < DIN / 32; ++kk) {
    const int k0 = kk * 32;
    ABCast au;
    au.q[0] = *(const uint4*)(Abase + k0 + kh * 16);       // K = k0+16kh .. +7
    au.q[1] = *(const uint4*)(Abase + k0 + 8 + kh * 16);   // K = k0+16kh+8 .. +7
    const v16bf a = au.v;
    const int krow = k0 + kh * 16;
    const v16bf b0 = load_b16(wx + (size_t)(0 * HID + h_base + nl) * DIN + krow);
    const v16bf b1 = load_b16(wx + (size_t)(1 * HID + h_base + nl) * DIN + krow);
    const v16bf b2 = load_b16(wx + (size_t)(2 * HID + h_base + nl) * DIN + krow);
    const v16bf b3 = load_b16(wx + (size_t)(3 * HID + h_base + nl) * DIN + krow);
    acc0 = __builtin_amdgcn_wmma_f32_16x16x32_bf16(false, a, false, b0, (short)0, acc0, false, false);
    acc1 = __builtin_amdgcn_wmma_f32_16x16x32_bf16(false, a, false, b1, (short)0, acc1, false, false);
    acc2 = __builtin_amdgcn_wmma_f32_16x16x32_bf16(false, a, false, b2, (short)0, acc2, false, false);
    acc3 = __builtin_amdgcn_wmma_f32_16x16x32_bf16(false, a, false, b3, (short)0, acc3, false, false);
  }

  // ---- K loop over recurrence (Wh, K = 0..1023) ----
  for (int kk = 0; kk < HID / 32; ++kk) {
    const int k0 = kk * 32;
    ABCast au;
    au.q[0] = *(const uint4*)(Abase + DIN + k0 + kh * 16);
    au.q[1] = *(const uint4*)(Abase + DIN + k0 + 8 + kh * 16);
    const v16bf a = au.v;
    const int krow = k0 + kh * 16;
    const v16bf b0 = load_b16(wh + (size_t)(0 * HID + h_base + nl) * HID + krow);
    const v16bf b1 = load_b16(wh + (size_t)(1 * HID + h_base + nl) * HID + krow);
    const v16bf b2 = load_b16(wh + (size_t)(2 * HID + h_base + nl) * HID + krow);
    const v16bf b3 = load_b16(wh + (size_t)(3 * HID + h_base + nl) * HID + krow);
    acc0 = __builtin_amdgcn_wmma_f32_16x16x32_bf16(false, a, false, b0, (short)0, acc0, false, false);
    acc1 = __builtin_amdgcn_wmma_f32_16x16x32_bf16(false, a, false, b1, (short)0, acc1, false, false);
    acc2 = __builtin_amdgcn_wmma_f32_16x16x32_bf16(false, a, false, b2, (short)0, acc2, false, false);
    acc3 = __builtin_amdgcn_wmma_f32_16x16x32_bf16(false, a, false, b3, (short)0, acc3, false, false);
  }

  // ---- Fused LSTM cell update ----
  // C/D layout: lanes 0-15: M = r; lanes 16-31: M = 8 + r; N = lane & 15.
  const int col = h_base + nl;
  const float bi = bias[0 * HID + col];
  const float bf = bias[1 * HID + col] + 1.0f;   // FORGET_BIAS folded in
  const float bj = bias[2 * HID + col];
  const float bo = bias[3 * HID + col];

#pragma unroll
  for (int r = 0; r < 8; ++r) {
    const int row = m_base + kh * 8 + r;
    const size_t idx = (size_t)row * HID + col;
    const float iv = sigm(acc0[r] + bi);
    const float fv = sigm(acc1[r] + bf);
    const float jv = tanhf(acc2[r] + bj);
    const float ov = sigm(acc3[r] + bo);
    const float cn = cst[idx] * fv + iv * jv;
    cst[idx]   = cn;
    hnext[idx] = tanhf(cn) * ov;
  }
}

// ---------------------------------------------------------------------------
// Finalize: d_out = concat(h_final, c_final)
// ---------------------------------------------------------------------------
__global__ __launch_bounds__(256) void lstm_fin_kernel(
    const float* __restrict__ h, const float* __restrict__ c,
    float* __restrict__ out)
{
  const int i = blockIdx.x * 256 + threadIdx.x;
  if (i < BATCH * HID) {
    out[i]               = h[i];
    out[BATCH * HID + i] = c[i];
  }
}

// ---------------------------------------------------------------------------
extern "C" void kernel_launch(void* const* d_in, const int* in_sizes, int n_in,
                              void* d_out, int out_size, void* d_ws, size_t ws_size,
                              hipStream_t stream) {
  const float* x    = (const float*)d_in[0];   // (512, 64, 512)
  const float* h0   = (const float*)d_in[1];   // (64, 1024)
  const float* c0   = (const float*)d_in[2];   // (64, 1024)
  const float* wx_w = (const float*)d_in[3];   // (4096, 512)
  const float* wh_w = (const float*)d_in[4];   // (4096, 1024)
  const float* wh_b = (const float*)d_in[5];   // (4096,)
  float* out = (float*)d_out;

  // Workspace layout (~12.8 MB total)
  char* ws = (char*)d_ws;
  unsigned short* wxb = (unsigned short*)ws;                       // 4 MB bf16
  unsigned short* whb = (unsigned short*)(ws + (size_t)4194304);   // 8 MB bf16
  float* hbuf = (float*)(ws + (size_t)4194304 + 8388608);          // 2 x 64x1024
  float* cbuf = hbuf + 2 * BATCH * HID;                            // 64x1024

  // 1) Prep: bf16 weights + seed state
  {
    const int n = G4 * HID;  // largest conversion (4,194,304)
    lstm_prep_kernel<<<(n + 255) / 256, 256, 0, stream>>>(
        wx_w, wh_w, h0, c0, wxb, whb, hbuf, cbuf);
  }

  // 2) 512 serial steps; kernel boundary = grid-wide sync on h
  for (int s = 0; s < SEQ; ++s) {
    const float* xs    = x + (size_t)s * BATCH * DIN;
    const float* hprev = hbuf + (size_t)(s & 1) * BATCH * HID;
    float*       hnext = hbuf + (size_t)((s + 1) & 1) * BATCH * HID;
    lstm_step_kernel<<<dim3(8, 4), 256, 0, stream>>>(
        xs, wxb, whb, wh_b, hprev, hnext, cbuf);
  }

  // 3) Finalize: h ends in hbuf[SEQ & 1 == 0]
  lstm_fin_kernel<<<(BATCH * HID + 255) / 256, 256, 0, stream>>>(
      hbuf, cbuf, out);
}
